// Bottleneck_Transformer_9002251453133
// MI455X (gfx1250) — matrix operand order repaired and checked by reading, not compiled, not run on hardware
//
#include <hip/hip_runtime.h>
#include <hip/hip_bf16.h>

typedef __bf16 bf16_t;
typedef __attribute__((ext_vector_type(16))) __bf16 v16bf;
typedef __attribute__((ext_vector_type(8)))  __bf16 v8bf;
typedef __attribute__((ext_vector_type(8)))  float  v8f;

#define BN_EPS 1e-5f

// ---------------------------------------------------------------------------
// Prep kernels
// ---------------------------------------------------------------------------

// Fold eval-mode BN into a 1x1-conv weight: W' = s*W (bf16), bias' = beta + s*(cb - mean)
// g==nullptr -> plain convert, bias' = cb.
__global__ __launch_bounds__(256) void fold_weight_kernel(
    const float* __restrict__ W, const float* __restrict__ g,
    const float* __restrict__ beta, const float* __restrict__ mean,
    const float* __restrict__ var, const float* __restrict__ convb,
    bf16_t* __restrict__ Wo, float* __restrict__ biasOut,
    int rows, int cols)
{
  long i = (long)blockIdx.x * 256 + threadIdx.x;
  long total = (long)rows * cols;
  if (i >= total) return;
  int r = (int)(i / cols);
  float s = g ? g[r] * rsqrtf(var[r] + BN_EPS) : 1.0f;
  Wo[i] = (bf16_t)(W[i] * s);
  if ((i % cols) == 0) {
    float cb = convb ? convb[r] : 0.0f;
    biasOut[r] = g ? (beta[r] + s * (cb - mean[r])) : cb;
  }
}

// bn2 is applied to an activation (not foldable into a weight): scale/bias vectors.
__global__ __launch_bounds__(256) void bn_fold_kernel(
    const float* __restrict__ g, const float* __restrict__ beta,
    const float* __restrict__ mean, const float* __restrict__ var,
    float* __restrict__ s, float* __restrict__ b, int n)
{
  int i = blockIdx.x * 256 + threadIdx.x;
  if (i >= n) return;
  float sc = g[i] * rsqrtf(var[i] + BN_EPS);
  s[i] = sc;
  b[i] = beta[i] - sc * mean[i];
}

// pos[h][d][n] = rel_h[h,d,hi] + rel_w[h,d,wi], n = wi*16 + hi  (bf16)
__global__ __launch_bounds__(256) void build_pos_kernel(
    const float* __restrict__ rel_h, const float* __restrict__ rel_w,
    bf16_t* __restrict__ pos)
{
  int i = blockIdx.x * 256 + threadIdx.x;   // 4*128*256
  if (i >= 4 * 128 * 256) return;
  int n  = i & 255;
  int hd = i >> 8;
  int wi = n >> 4, hi = n & 15;
  pos[i] = (bf16_t)(rel_h[hd * 16 + hi] + rel_w[hd * 16 + wi]);
}

__global__ __launch_bounds__(256) void f2bf_kernel(
    const float* __restrict__ x, bf16_t* __restrict__ y, long n)
{
  long i = (long)blockIdx.x * 256 + threadIdx.x;
  if (i < n) y[i] = (bf16_t)x[i];
}

// ---------------------------------------------------------------------------
// Generic bf16 WMMA GEMM:  C[z, m, n] = sum_k opA(A)[m,k] * opB(B)[k,n]  (+ epilogue)
//   TA: A given as [K x M] (read A[k*lda + m]);  else A is [M x K] (A[m*lda + k])
//   TB: B given as [N x K] (read B[n*ldb + k]);  else B is [K x N] (B[k*ldb + n])
// EPI: 0 none | 1 +bias | 2 relu(+bias) | 3 relu(scale*v+bias) | 4 +Add | 5 relu(+bias+Add)
// OBF: store bf16 else f32.  ch = sbMod ? (z%sbMod)*M + m : m.
// Macro tile 128x128, BK=32, 256 threads = 8 waves (4 along M x 2 along N),
// each wave computes 2x4 fragments of 16x16 via v_wmma_f32_16x16x32_bf16.
// When neither operand is transposed, global->LDS staging uses double-buffered
// GLOBAL_LOAD_ASYNC_TO_LDS_B128 (ASYNCcnt) so the copy engine overlaps WMMA.
// ---------------------------------------------------------------------------
template<bool TA, bool TB, int EPI, bool OBF>
__global__ __launch_bounds__(256) void gemm_wmma_kernel(
    const bf16_t* __restrict__ Ag, const bf16_t* __restrict__ Bg,
    void* __restrict__ Cg, const float* __restrict__ Addg,
    const float* __restrict__ scale, const float* __restrict__ bias,
    int M, int N, int K, int lda, int ldb, int ldc,
    long strideA, long strideB, long strideC, long strideAdd,
    int aMod, int sbMod)
{
  constexpr bool ASYNC = (!TA) && (!TB);

  __shared__ __align__(16) bf16_t sA[2][128 * 32];   // [buf][m][k] row-major
  __shared__ __align__(16) bf16_t sB[2][32 * 128];   // [buf][k][n] row-major

  const int tid  = threadIdx.x;
  const int lane = tid & 31;
  const int wave = tid >> 5;
  const int wM   = wave & 3;    // 4 waves along M (32 rows each)
  const int wN   = wave >> 2;   // 2 waves along N (64 cols each)

  const int z  = blockIdx.z;
  const int m0 = blockIdx.y * 128;
  const int n0 = blockIdx.x * 128;

  const bf16_t* A = Ag + (long)(aMod ? (z % aMod) : z) * strideA;
  const bf16_t* B = Bg + (long)z * strideB;

  // ---- per-thread chunk coordinates for the 128x32 / 32x128 tiles ----
  const int c0 = tid, c1 = tid + 256;
  const int mmA0 = c0 >> 2,  kkA0 = (c0 & 3) << 3;
  const int mmA1 = c1 >> 2,  kkA1 = (c1 & 3) << 3;
  const int kkB0 = c0 >> 4,  nnB0 = (c0 & 15) << 3;
  const int kkB1 = c1 >> 4,  nnB1 = (c1 & 15) << 3;

  // ---- async-copy machinery (ASYNC instantiations only) ----
  const bf16_t* gA0 = &A[(long)(m0 + mmA0) * lda + kkA0];
  const bf16_t* gA1 = &A[(long)(m0 + mmA1) * lda + kkA1];
  const bf16_t* gB0 = &B[(long)kkB0 * ldb + n0 + nnB0];
  const bf16_t* gB1 = &B[(long)kkB1 * ldb + n0 + nnB1];
  const unsigned lA0 = (unsigned)(size_t)&sA[0][mmA0 * 32 + kkA0];
  const unsigned lA1 = (unsigned)(size_t)&sA[0][mmA1 * 32 + kkA1];
  const unsigned lB0 = (unsigned)(size_t)&sB[0][kkB0 * 128 + nnB0];
  const unsigned lB1 = (unsigned)(size_t)&sB[0][kkB1 * 128 + nnB1];
  const long bStepB = (long)32 * ldb;   // B advances 32 K-rows per tile

  auto issue_async = [&](int buf) {
    const unsigned bo = buf ? 8192u : 0u;   // sizeof(sA[0]) == sizeof(sB[0]) == 8192 B
    asm volatile("global_load_async_to_lds_b128 %0, %1, off"
                 :: "v"(lA0 + bo), "v"(gA0) : "memory");
    asm volatile("global_load_async_to_lds_b128 %0, %1, off"
                 :: "v"(lA1 + bo), "v"(gA1) : "memory");
    asm volatile("global_load_async_to_lds_b128 %0, %1, off"
                 :: "v"(lB0 + bo), "v"(gB0) : "memory");
    asm volatile("global_load_async_to_lds_b128 %0, %1, off"
                 :: "v"(lB1 + bo), "v"(gB1) : "memory");
    gA0 += 32; gA1 += 32; gB0 += bStepB; gB1 += bStepB;
  };

  v8f acc[2][4] = {};
  int buf = 0;
  if constexpr (ASYNC) issue_async(0);

  for (int k0 = 0; k0 < K; k0 += 32) {
    if constexpr (ASYNC) {
      // Double-buffered: kick off tile k0+32 into buf^1, then wait for tile k0.
      // Async loads complete in order, so ASYNCcnt<=4 => previous 4 are done.
      if (k0 + 32 < K) {
        issue_async(buf ^ 1);
        asm volatile("s_wait_asynccnt 4" ::: "memory");
      } else {
        asm volatile("s_wait_asynccnt 0" ::: "memory");
      }
    } else {
      // ---- VGPR-staged path for transposed operands (single buffer 0) ----
      if (!TA) {
        #pragma unroll
        for (int i = 0; i < 2; ++i) {
          int c  = tid + 256 * i;
          int mm = c >> 2;
          int kk = (c & 3) << 3;
          *(v8bf*)&sA[0][mm * 32 + kk] =
              *(const v8bf*)&A[(long)(m0 + mm) * lda + k0 + kk];
        }
      } else {
        #pragma unroll
        for (int i = 0; i < 2; ++i) {
          int c  = tid + 256 * i;
          int kk = c >> 4;
          int mm = (c & 15) << 3;
          v8bf t = *(const v8bf*)&A[(long)(k0 + kk) * lda + m0 + mm];
          #pragma unroll
          for (int e = 0; e < 8; ++e) sA[0][(mm + e) * 32 + kk] = t[e];
        }
      }
      if (!TB) {
        #pragma unroll
        for (int i = 0; i < 2; ++i) {
          int c  = tid + 256 * i;
          int kk = c >> 4;
          int nn = (c & 15) << 3;
          *(v8bf*)&sB[0][kk * 128 + nn] =
              *(const v8bf*)&B[(long)(k0 + kk) * ldb + n0 + nn];
        }
      } else {
        #pragma unroll
        for (int i = 0; i < 2; ++i) {
          int c  = tid + 256 * i;
          int nn = c >> 2;
          int kk = (c & 3) << 3;
          v8bf t = *(const v8bf*)&B[(long)(n0 + nn) * ldb + k0 + kk];
          #pragma unroll
          for (int e = 0; e < 8; ++e) sB[0][(kk + e) * 128 + nn] = t[e];
        }
      }
      // L2 prefetch of next K tile
      if (k0 + 32 < K) {
        if (!TA) __builtin_prefetch(&A[(long)(m0 + (tid >> 1)) * lda + k0 + 32], 0, 1);
        else     __builtin_prefetch(&A[(long)(k0 + 32 + (tid & 31)) * lda + m0 + ((tid >> 5) << 4)], 0, 1);
        if (!TB) __builtin_prefetch(&B[(long)(k0 + 32 + (tid & 31)) * ldb + n0 + ((tid >> 5) << 4)], 0, 1);
        else     __builtin_prefetch(&B[(long)(n0 + (tid >> 1)) * ldb + k0 + 32], 0, 1);
      }
    }
    __syncthreads();

    // ---- fragments + WMMA ----
    {
      const bf16_t* sAb = &sA[ASYNC ? buf : 0][0];
      const bf16_t* sBb = &sB[ASYNC ? buf : 0][0];
      const int mRow = lane & 15;
      const int hk   = lane >> 4;
      v16bf af[2];
      #pragma unroll
      for (int fm = 0; fm < 2; ++fm) {
        const bf16_t* r = &sAb[(wM * 32 + fm * 16 + mRow) * 32 + hk * 8];
        #pragma unroll
        for (int j = 0; j < 8; ++j)  af[fm][j] = r[j];       // K = j + 8*hk
        #pragma unroll
        for (int j = 8; j < 16; ++j) af[fm][j] = r[j + 8];   // K = j + 8 + 8*hk
      }
      v16bf bfr[4];
      #pragma unroll
      for (int fn = 0; fn < 4; ++fn) {
        const bf16_t* r = &sBb[(hk * 8) * 128 + wN * 64 + fn * 16 + mRow];
        #pragma unroll
        for (int j = 0; j < 8; ++j)  bfr[fn][j] = r[j * 128];
        #pragma unroll
        for (int j = 8; j < 16; ++j) bfr[fn][j] = r[(j + 8) * 128];
      }
      #pragma unroll
      for (int fm = 0; fm < 2; ++fm)
        #pragma unroll
        for (int fn = 0; fn < 4; ++fn)
          acc[fm][fn] = __builtin_amdgcn_wmma_f32_16x16x32_bf16(
              false, af[fm], false, bfr[fn], (short)0, acc[fm][fn], false, false);
    }
    __syncthreads();   // all waves done reading buf before it is refilled
    buf ^= 1;
  }

  // ---- epilogue: C/D layout: lanes 0-15 N=lane (M=j), lanes 16-31 (M=j+8) ----
  const int hl = lane >> 4;
  const int ln = lane & 15;
  float*  Cf = (float*)Cg;
  bf16_t* Cb = (bf16_t*)Cg;
  const long coff   = (long)z * strideC;
  const long aoff   = (long)z * strideAdd;
  const int  chBase = sbMod ? ((z % sbMod) * M) : 0;
  #pragma unroll
  for (int fm = 0; fm < 2; ++fm) {
    #pragma unroll
    for (int fn = 0; fn < 4; ++fn) {
      #pragma unroll
      for (int j = 0; j < 8; ++j) {
        int m = m0 + wM * 32 + fm * 16 + j + 8 * hl;
        int n = n0 + wN * 64 + fn * 16 + ln;
        float v = acc[fm][fn][j];
        int ch = chBase + m;
        if constexpr (EPI == 1) { v += bias[ch]; }
        else if constexpr (EPI == 2) { v += bias[ch]; v = fmaxf(v, 0.0f); }
        else if constexpr (EPI == 3) { v = v * scale[ch] + bias[ch]; v = fmaxf(v, 0.0f); }
        else if constexpr (EPI == 4) { v += Addg[aoff + (long)m * ldc + n]; }
        else if constexpr (EPI == 5) { v += bias[ch] + Addg[aoff + (long)m * ldc + n]; v = fmaxf(v, 0.0f); }
        long ci = coff + (long)m * ldc + n;
        if constexpr (OBF) Cb[ci] = (bf16_t)v; else Cf[ci] = v;
      }
    }
  }
}

// ---------------------------------------------------------------------------
// Row softmax: one wave32 per 256-wide row; wave shuffle reductions.
// ---------------------------------------------------------------------------
__global__ __launch_bounds__(256) void softmax_rows_kernel(
    const float* __restrict__ S, bf16_t* __restrict__ P, long nRows)
{
  long row = (long)blockIdx.x * 8 + (threadIdx.x >> 5);
  if (row >= nRows) return;
  int lane = threadIdx.x & 31;
  const float* r = S + row * 256;
  float x[8];
  float mx = -3.4e38f;
  #pragma unroll
  for (int i = 0; i < 8; ++i) { x[i] = r[lane + 32 * i]; mx = fmaxf(mx, x[i]); }
  #pragma unroll
  for (int off = 16; off > 0; off >>= 1) mx = fmaxf(mx, __shfl_xor(mx, off, 32));
  float sum = 0.0f;
  #pragma unroll
  for (int i = 0; i < 8; ++i) { x[i] = __expf(x[i] - mx); sum += x[i]; }
  #pragma unroll
  for (int off = 16; off > 0; off >>= 1) sum += __shfl_xor(sum, off, 32);
  float inv = 1.0f / sum;
  bf16_t* o = P + row * 256;
  #pragma unroll
  for (int i = 0; i < 8; ++i) o[lane + 32 * i] = (bf16_t)(x[i] * inv);
}

// ---------------------------------------------------------------------------
// Host launcher
// ---------------------------------------------------------------------------
extern "C" void kernel_launch(void* const* d_in, const int* in_sizes, int n_in,
                              void* d_out, int out_size, void* d_ws, size_t ws_size,
                              hipStream_t stream) {
  // Problem constants
  const int Bn = 64, Cin = 1024, P = 512, Cout = 2048, N = 256, HEADS = 4, D = 128;
  const int BH = Bn * HEADS;           // 256

  const float* x       = (const float*)d_in[0];
  const float* conv1_w = (const float*)d_in[1];
  const float* bn1_g   = (const float*)d_in[2];
  const float* bn1_b   = (const float*)d_in[3];
  const float* bn1_m   = (const float*)d_in[4];
  const float* bn1_v   = (const float*)d_in[5];
  const float* q_w     = (const float*)d_in[6];
  const float* q_b     = (const float*)d_in[7];
  const float* k_w     = (const float*)d_in[8];
  const float* k_b     = (const float*)d_in[9];
  const float* v_w     = (const float*)d_in[10];
  const float* v_b     = (const float*)d_in[11];
  const float* rel_h   = (const float*)d_in[12];
  const float* rel_w   = (const float*)d_in[13];
  const float* bn2_g   = (const float*)d_in[14];
  const float* bn2_b   = (const float*)d_in[15];
  const float* bn2_m   = (const float*)d_in[16];
  const float* bn2_v   = (const float*)d_in[17];
  const float* conv3_w = (const float*)d_in[18];
  const float* bn3_g   = (const float*)d_in[19];
  const float* bn3_b   = (const float*)d_in[20];
  const float* bn3_m   = (const float*)d_in[21];
  const float* bn3_v   = (const float*)d_in[22];
  const float* sc_w    = (const float*)d_in[23];
  const float* sc_b    = (const float*)d_in[24];
  const float* scbn_g  = (const float*)d_in[25];
  const float* scbn_b  = (const float*)d_in[26];
  const float* scbn_m  = (const float*)d_in[27];
  const float* scbn_v  = (const float*)d_in[28];

  // ---- workspace carve (256B aligned slabs) ----
  char* base = (char*)d_ws;
  size_t off = 0;
  auto carve = [&](size_t bytes) -> void* {
    void* p = base + off;
    off = (off + bytes + 255) & ~(size_t)255;
    return p;
  };
  bf16_t* W1b  = (bf16_t*)carve((size_t)P * Cin * 2);
  float*  b1   = (float*) carve((size_t)P * 4);
  bf16_t* Wqb  = (bf16_t*)carve((size_t)P * P * 2);
  float*  bq   = (float*) carve((size_t)P * 4);
  bf16_t* Wkb  = (bf16_t*)carve((size_t)P * P * 2);
  float*  bk   = (float*) carve((size_t)P * 4);
  bf16_t* Wvb  = (bf16_t*)carve((size_t)P * P * 2);
  float*  bv   = (float*) carve((size_t)P * 4);
  bf16_t* W3b  = (bf16_t*)carve((size_t)Cout * P * 2);
  float*  b3   = (float*) carve((size_t)Cout * 4);
  bf16_t* WSb  = (bf16_t*)carve((size_t)Cout * Cin * 2);
  float*  bsc  = (float*) carve((size_t)Cout * 4);
  bf16_t* posb = (bf16_t*)carve((size_t)HEADS * D * N * 2);
  float*  s2   = (float*) carve((size_t)P * 4);
  float*  b2   = (float*) carve((size_t)P * 4);
  bf16_t* Xb   = (bf16_t*)carve((size_t)Bn * Cin * N * 2);
  bf16_t* Y1   = (bf16_t*)carve((size_t)Bn * P * N * 2);
  bf16_t* Qm   = (bf16_t*)carve((size_t)Bn * P * N * 2);
  bf16_t* Km   = (bf16_t*)carve((size_t)Bn * P * N * 2);
  bf16_t* Vm   = (bf16_t*)carve((size_t)Bn * P * N * 2);
  float*  Sm   = (float*) carve((size_t)BH * N * N * 4);
  bf16_t* Pm   = (bf16_t*)carve((size_t)BH * N * N * 2);
  bf16_t* Y2   = (bf16_t*)carve((size_t)Bn * P * N * 2);
  (void)ws_size;

  // ---- prep: fold BN into weights, build pos, convert x ----
  fold_weight_kernel<<<(P * Cin + 255) / 256, 256, 0, stream>>>(
      conv1_w, bn1_g, bn1_b, bn1_m, bn1_v, nullptr, W1b, b1, P, Cin);
  fold_weight_kernel<<<(P * P + 255) / 256, 256, 0, stream>>>(
      q_w, nullptr, nullptr, nullptr, nullptr, q_b, Wqb, bq, P, P);
  fold_weight_kernel<<<(P * P + 255) / 256, 256, 0, stream>>>(
      k_w, nullptr, nullptr, nullptr, nullptr, k_b, Wkb, bk, P, P);
  fold_weight_kernel<<<(P * P + 255) / 256, 256, 0, stream>>>(
      v_w, nullptr, nullptr, nullptr, nullptr, v_b, Wvb, bv, P, P);
  fold_weight_kernel<<<(Cout * P + 255) / 256, 256, 0, stream>>>(
      conv3_w, bn3_g, bn3_b, bn3_m, bn3_v, nullptr, W3b, b3, Cout, P);
  fold_weight_kernel<<<(Cout * Cin + 255) / 256, 256, 0, stream>>>(
      sc_w, scbn_g, scbn_b, scbn_m, scbn_v, sc_b, WSb, bsc, Cout, Cin);
  bn_fold_kernel<<<(P + 255) / 256, 256, 0, stream>>>(
      bn2_g, bn2_b, bn2_m, bn2_v, s2, b2, P);
  build_pos_kernel<<<(HEADS * D * N + 255) / 256, 256, 0, stream>>>(rel_h, rel_w, posb);
  {
    long nx = (long)Bn * Cin * N;
    f2bf_kernel<<<(unsigned)((nx + 255) / 256), 256, 0, stream>>>(x, Xb, nx);
  }

  // ---- 1) Y1 = relu(W1' @ X + b1')  [B,512,256] bf16 ----
  gemm_wmma_kernel<false, false, 2, true><<<dim3(2, 4, Bn), 256, 0, stream>>>(
      W1b, Xb, Y1, nullptr, nullptr, b1,
      P, N, Cin, Cin, N, N,
      0L, (long)Cin * N, (long)P * N, 0L, 0, 0);

  // ---- 2) Q/K/V = Wq/k/v @ Y1 + bias  [B,512,256] bf16 ----
  gemm_wmma_kernel<false, false, 1, true><<<dim3(2, 4, Bn), 256, 0, stream>>>(
      Wqb, Y1, Qm, nullptr, nullptr, bq,
      P, N, P, P, N, N, 0L, (long)P * N, (long)P * N, 0L, 0, 0);
  gemm_wmma_kernel<false, false, 1, true><<<dim3(2, 4, Bn), 256, 0, stream>>>(
      Wkb, Y1, Km, nullptr, nullptr, bk,
      P, N, P, P, N, N, 0L, (long)P * N, (long)P * N, 0L, 0, 0);
  gemm_wmma_kernel<false, false, 1, true><<<dim3(2, 4, Bn), 256, 0, stream>>>(
      Wvb, Y1, Vm, nullptr, nullptr, bv,
      P, N, P, P, N, N, 0L, (long)P * N, (long)P * N, 0L, 0, 0);

  // ---- 3) S = Q^T K  per (b,h): M=N=256, K=128 ----
  gemm_wmma_kernel<true, false, 0, false><<<dim3(2, 2, BH), 256, 0, stream>>>(
      Qm, Km, Sm, nullptr, nullptr, nullptr,
      N, N, D, N, N, N,
      (long)D * N, (long)D * N, (long)N * N, 0L, 0, 0);

  // ---- 4) S += pos^T Q  (pos indexed by head only: aMod=4) ----
  gemm_wmma_kernel<true, false, 4, false><<<dim3(2, 2, BH), 256, 0, stream>>>(
      posb, Qm, Sm, Sm, nullptr, nullptr,
      N, N, D, N, N, N,
      (long)D * N, (long)D * N, (long)N * N, (long)N * N, HEADS, 0);

  // ---- 5) P = softmax_rows(S)  (bf16) ----
  {
    long rows = (long)BH * N;                 // 65536
    softmax_rows_kernel<<<(unsigned)(rows / 8), 256, 0, stream>>>(Sm, Pm, rows);
  }

  // ---- 6) Y2 = relu(s2 * (V @ P^T) + b2)  per (b,h): M=128, N=256, K=256 ----
  gemm_wmma_kernel<false, true, 3, true><<<dim3(2, 1, BH), 256, 0, stream>>>(
      Vm, Pm, Y2, nullptr, s2, b2,
      D, N, N, N, N, N,
      (long)D * N, (long)N * N, (long)D * N, 0L, 0, HEADS);

  // ---- 7) d_out = WS' @ X + bsc'  (shortcut, f32) ----
  gemm_wmma_kernel<false, false, 1, false><<<dim3(2, 16, Bn), 256, 0, stream>>>(
      WSb, Xb, d_out, nullptr, nullptr, bsc,
      Cout, N, Cin, Cin, N, N,
      0L, (long)Cin * N, (long)Cout * N, 0L, 0, 0);

  // ---- 8) d_out = relu(W3' @ Y2 + b3' + d_out) ----
  gemm_wmma_kernel<false, false, 5, false><<<dim3(2, 16, Bn), 256, 0, stream>>>(
      W3b, Y2, d_out, (const float*)d_out, nullptr, b3,
      Cout, N, P, P, N, N,
      0L, (long)P * N, (long)Cout * N, (long)Cout * N, 0, 0);

  (void)in_sizes; (void)n_in; (void)out_size;
}
